// RNN_65163243814977
// MI455X (gfx1250) — matrix operand order repaired and checked by reading, not compiled, not run on hardware
//
#include <hip/hip_runtime.h>

#define B_TOTAL 32768
#define T_STEPS 28
#define IN_DIM  28
#define HID     64
#define ODIM    10

typedef __attribute__((ext_vector_type(16))) _Float16     v16h;
typedef __attribute__((ext_vector_type(8)))  float        v8f;
typedef __attribute__((ext_vector_type(4)))  float        v4f;
typedef __attribute__((ext_vector_type(8)))  unsigned int v8u;

__device__ __forceinline__ float fast_tanh(float x) {
#if __has_builtin(__builtin_amdgcn_tanhf)
    return __builtin_amdgcn_tanhf(x);          // v_tanh_f32 (TRANS32)
#elif __has_builtin(__builtin_amdgcn_tanh_f32)
    return __builtin_amdgcn_tanh_f32(x);
#else
    float r;
    asm("v_tanh_f32 %0, %1" : "=v"(r) : "v"(x));
    return r;
#endif
}

// lane <-> lane^16 exchange (SWAPX16: xor=0x10,or=0,and=0x1f -> 0x401f)
__device__ __forceinline__ unsigned swap16(unsigned v) {
#if __has_builtin(__builtin_amdgcn_ds_swizzle)
    return (unsigned)__builtin_amdgcn_ds_swizzle((int)v, 0x401f);
#else
    return (unsigned)__shfl_xor((int)v, 16, 32);
#endif
}

__device__ __forceinline__ unsigned pack2h(float a, float b) {
    union { _Float16 h[2]; unsigned u; } x;
    x.h[0] = (_Float16)a; x.h[1] = (_Float16)b;   // -> v_cvt_pk_f16_f32
    return x.u;
}

__global__ __launch_bounds__(128) void rnn_fused_kernel(
    const float* __restrict__ X,     // [B, T, 28]
    const float* __restrict__ W_ih,  // [64, 28]
    const float* __restrict__ W_hh,  // [64, 64]
    const float* __restrict__ b_ih,  // [64]
    const float* __restrict__ b_hh,  // [64]
    const float* __restrict__ W_out, // [10, 64]
    const float* __restrict__ b_out, // [10]
    float* __restrict__ out)         // [B, 10]
{
    __shared__ __align__(16) float hlast[4][16][HID];   // per-wave final h (f32)
    __shared__ float wout_s[ODIM * HID];
    __shared__ float bout_s[ODIM];

    const int tid  = threadIdx.x;
    const int wid  = tid >> 5;
    const int lane = tid & 31;
    const int nl   = lane & 15;     // batch row within tile (N for B/C operands)
    const int hi   = lane >> 4;     // half-wave select
    const int b0   = (blockIdx.x * 4 + wid) * 16;

    for (int i = tid; i < ODIM * HID; i += 128) wout_s[i] = W_out[i];
    if (tid < ODIM) bout_s[tid] = b_out[tid];
    __syncthreads();

    // ---- Loop-invariant A-operands: weight tiles, M = hidden rows.
    // A layout (16-bit, 16x32): lane row M = nl; halves 0..7 = K(hi*16..+7),
    // halves 8..15 = K(hi*16+8..+7) per 32-wide K step.
    // Bias trick: K=28 of the x-proj GEMM is a pad slot; put (b_ih+b_hh)[n]
    // there (A half index 12 on hi=1 lanes) and feed B_x[k=28] = 1.0, so the
    // accumulator can start from inline 0 (no bias registers, no C init).
    v16h Aih[4];        // W_ih tiles  (K = 0..27, K=28 = bias column)
    v16h Ahh[4][2];     // W_hh tiles  (K steps 0,1)
    #pragma unroll
    for (int nt = 0; nt < 4; ++nt) {
        const int n = nt * 16 + nl;
        const float bsum = b_ih[n] + b_hh[n];
        #pragma unroll
        for (int j = 0; j < 8; ++j) {
            const int k1 = hi * 16 + j;         // <= 23, always valid
            const int k2 = hi * 16 + 8 + j;     // hi=1: 24..31 tail
            _Float16 t2;
            if (k2 < IN_DIM)       t2 = (_Float16)W_ih[n * IN_DIM + k2];
            else if (k2 == IN_DIM) t2 = (_Float16)bsum;   // bias column
            else                   t2 = (_Float16)0.0f;
            Aih[nt][j]     = (_Float16)W_ih[n * IN_DIM + k1];
            Aih[nt][8 + j] = t2;
        }
        #pragma unroll
        for (int ks = 0; ks < 2; ++ks) {
            #pragma unroll
            for (int j = 0; j < 8; ++j) {
                Ahh[nt][ks][j]     = (_Float16)W_hh[n * HID + ks * 32 + hi * 16 + j];
                Ahh[nt][ks][8 + j] = (_Float16)W_hh[n * HID + ks * 32 + hi * 16 + 8 + j];
            }
        }
    }

    // ---- B-operand h^T fragments, start at h0 = 0.
    v16h Bh[2];
    #pragma unroll
    for (int j = 0; j < 16; ++j) { Bh[0][j] = (_Float16)0.0f; Bh[1][j] = (_Float16)0.0f; }

    const float* xrow = X + (size_t)(b0 + nl) * (T_STEPS * IN_DIM);
    float ht[4][8];   // tanh outputs; after loop = final h for this lane

    #pragma unroll 1
    for (int t = 0; t < T_STEPS; ++t) {
        // ---- B-operand X^T: lane = batch row nl, K = hi*16 .. hi*16+15,
        // contiguous in memory. hi=1 tail (K=28..31) = {1,0,0,0} bias column;
        // branchless: always-valid load + cndmask selects (no EXEC juggling).
        const float* xr = xrow + t * IN_DIM;
        v4f x0 = *(const v4f*)(xr + hi * 16);
        v4f x1 = *(const v4f*)(xr + hi * 16 + 4);
        v4f x2 = *(const v4f*)(xr + hi * 16 + 8);
        v4f x3l = *(const v4f*)(xr + (hi ? 0 : 12));   // hi=1: dummy valid addr
        v4f x3;
        x3[0] = hi ? 1.0f : x3l[0];
        x3[1] = hi ? 0.0f : x3l[1];
        x3[2] = hi ? 0.0f : x3l[2];
        x3[3] = hi ? 0.0f : x3l[3];
        v16h Bx;
        #pragma unroll
        for (int j = 0; j < 4; ++j) {
            Bx[j]      = (_Float16)x0[j];
            Bx[4 + j]  = (_Float16)x1[j];
            Bx[8 + j]  = (_Float16)x2[j];
            Bx[12 + j] = (_Float16)x3[j];
        }

        // ---- 12 WMMAs: D[n][b] = (bias|W_ih)·(x|1)^T + W_hh·h^T per tile
        #pragma unroll
        for (int nt = 0; nt < 4; ++nt) {
            v8f acc = {0.f, 0.f, 0.f, 0.f, 0.f, 0.f, 0.f, 0.f};
            acc = __builtin_amdgcn_wmma_f32_16x16x32_f16(
                false, Aih[nt], false, Bx, (short)0, acc, false, false);
            acc = __builtin_amdgcn_wmma_f32_16x16x32_f16(
                false, Ahh[nt][0], false, Bh[0], (short)0, acc, false, false);
            acc = __builtin_amdgcn_wmma_f32_16x16x32_f16(
                false, Ahh[nt][1], false, Bh[1], (short)0, acc, false, false);
            #pragma unroll
            for (int v = 0; v < 8; ++v) ht[nt][v] = fast_tanh(acc[v]);
        }

        // ---- Rebuild Bh for next step: pack pairs + half-wave swap.
        // Lane holds h[b=nl][n = nt*16 + 8*hi + v]; consecutive v = consecutive n.
        unsigned pk[4][4];
        #pragma unroll
        for (int nt = 0; nt < 4; ++nt)
            #pragma unroll
            for (int q = 0; q < 4; ++q)
                pk[nt][q] = pack2h(ht[nt][2 * q], ht[nt][2 * q + 1]);

        #pragma unroll
        for (int ks = 0; ks < 2; ++ks) {
            v8u bu;
            #pragma unroll
            for (int q = 0; q < 4; ++q) {
                // send the group our partner (lane^16) needs; receive ours
                const unsigned send = hi ? pk[2 * ks][q] : pk[2 * ks + 1][q];
                const unsigned recv = swap16(send);
                bu[q]     = hi ? recv              : pk[2 * ks][q];   // K lo half
                bu[4 + q] = hi ? pk[2 * ks + 1][q] : recv;            // K hi half
            }
            Bh[ks] = __builtin_bit_cast(v16h, bu);
        }
    }

    // ---- Stash final h (f32) to LDS for the head.
    #pragma unroll
    for (int nt = 0; nt < 4; ++nt) {
        v4f lo  = {ht[nt][0], ht[nt][1], ht[nt][2], ht[nt][3]};
        v4f hi4 = {ht[nt][4], ht[nt][5], ht[nt][6], ht[nt][7]};
        *(v4f*)&hlast[wid][nl][nt * 16 + 8 * hi]     = lo;
        *(v4f*)&hlast[wid][nl][nt * 16 + 8 * hi + 4] = hi4;
    }
    asm volatile("s_wait_dscnt 0x0" ::: "memory");

    // ---- Head: out[b, o] = h_last . W_out[o] + b_out[o]
    const int obase = hi * 5;
    #pragma unroll
    for (int oo = 0; oo < 5; ++oo) {
        const int o = obase + oo;
        float s = bout_s[o];
        #pragma unroll
        for (int k = 0; k < HID; ++k)
            s += hlast[wid][nl][k] * wout_s[o * HID + k];
        out[(size_t)(b0 + nl) * ODIM + o] = s;
    }
}

extern "C" void kernel_launch(void* const* d_in, const int* in_sizes, int n_in,
                              void* d_out, int out_size, void* d_ws, size_t ws_size,
                              hipStream_t stream) {
    const float* X     = (const float*)d_in[0];
    const float* W_ih  = (const float*)d_in[1];
    const float* W_hh  = (const float*)d_in[2];
    const float* b_ih  = (const float*)d_in[3];
    const float* b_hh  = (const float*)d_in[4];
    const float* W_out = (const float*)d_in[5];
    const float* b_out = (const float*)d_in[6];
    float* out = (float*)d_out;

    dim3 grid(B_TOTAL / 64);   // 16 rows/wave * 4 waves/block
    dim3 block(128);
    rnn_fused_kernel<<<grid, block, 0, stream>>>(X, W_ih, W_hh, b_ih, b_hh,
                                                 W_out, b_out, out);
}
